// SelfAttention_3917010174225
// MI455X (gfx1250) — matrix operand order repaired and checked by reading, not compiled, not run on hardware
//
#include <hip/hip_runtime.h>
#include <hip/hip_bf16.h>

// ---------------------------------------------------------------------------
// Self-attention (B=2, T=2048, C=768, H=12, hd=64) for gfx1250 (MI455X).
// bf16 WMMA (v_wmma_f32_16x16x32_bf16) + flash-attention structure +
// GLOBAL_LOAD_ASYNC_TO_LDS_B128 staging (ASYNCcnt path).
// ---------------------------------------------------------------------------

typedef __attribute__((ext_vector_type(16))) __bf16 v16bf;
typedef __attribute__((ext_vector_type(8)))  float  v8f;

#define Bn   2
#define Tn   2048
#define Cn   768
#define Hn   12
#define HD   64
#define Mtot (Bn * Tn)      // 4096

// ---------------- helpers ----------------

__device__ inline ushort f2bf(float x) {
    union { float f; unsigned u; } v; v.f = x;
    unsigned r = v.u + 0x7fffu + ((v.u >> 16) & 1u);   // round-to-nearest-even
    return (ushort)(r >> 16);
}

// Build a 16-element bf16 fragment from two 16-byte chunks (LDS or global).
__device__ inline v16bf frag2(const ushort* p0, const ushort* p1) {
    union { v16bf v; uint4 u[2]; } f;
    f.u[0] = *(const uint4*)p0;
    f.u[1] = *(const uint4*)p1;
    return f.v;
}

// Async copy of 16 bytes per lane: global memory -> LDS (ASYNCcnt-tracked).
// lds_off: byte offset within the workgroup's LDS allocation.
__device__ inline void async_cp16(unsigned lds_off, const void* gaddr) {
    asm volatile("global_load_async_to_lds_b128 %0, %1, off"
                 :: "v"(lds_off), "v"(gaddr) : "memory");
}
__device__ inline void wait_async0() {
    asm volatile("s_wait_asynccnt 0x0" ::: "memory");
}
__device__ inline unsigned lds_off32(const void* p) {
    return (unsigned)(size_t)p;   // low 32 bits of generic addr = LDS offset
}

__device__ inline float rmax16(float v) {
    #pragma unroll
    for (int i = 1; i < 16; i <<= 1) v = fmaxf(v, __shfl_xor(v, i, 32));
    return v;
}
__device__ inline float rsum16(float v) {
    #pragma unroll
    for (int i = 1; i < 16; i <<= 1) v += __shfl_xor(v, i, 32);
    return v;
}

// ---------------- kernel 1: x fp32 -> bf16 ----------------

__global__ void cvt_x_bf16(const float* __restrict__ x, ushort* __restrict__ xb) {
    int idx = (blockIdx.x * 256 + threadIdx.x) * 8;
    ushort tmp[8];
    #pragma unroll
    for (int i = 0; i < 8; ++i) tmp[i] = f2bf(x[idx + i]);
    *(uint4*)&xb[idx] = *(const uint4*)tmp;
}

// ---------------- kernel 2: W[o][c] fp32 -> Wt[c][o] bf16 (4 weights) ------

__global__ void cvt_w_bf16(const float* __restrict__ Wq, const float* __restrict__ Wk,
                           const float* __restrict__ Wv, const float* __restrict__ Wp,
                           ushort* __restrict__ wt) {
    int slot = blockIdx.z;
    const float* W = (slot == 0) ? Wq : (slot == 1) ? Wk : (slot == 2) ? Wv : Wp;
    __shared__ float tile[32][33];
    int o0 = blockIdx.x * 32, c0 = blockIdx.y * 32;
    int tx = threadIdx.x, ty = threadIdx.y;
    #pragma unroll
    for (int i = 0; i < 32; i += 8)
        tile[ty + i][tx] = W[(size_t)(o0 + ty + i) * Cn + c0 + tx];
    __syncthreads();
    ushort* dst = wt + (size_t)slot * Cn * Cn;
    #pragma unroll
    for (int i = 0; i < 32; i += 8)
        dst[(size_t)(c0 + ty + i) * Cn + o0 + tx] = f2bf(tile[tx][ty + i]);
}

// ---------------- kernel 3/5: GEMM  out[M,N] = A[M,K] * Wt[K,N] + bias -----
// modes: 0=Q -> [B,H,T,64] bf16, 1=K -> same, 2=V -> [B,H,64,T] bf16,
//        3=proj -> fp32 [M,C] (d_out)

#define BM 128
#define BN 64
#define BK 64
#define SA 80   // LDS halfword stride (non-pow2, 16B-aligned rows)

__global__ __launch_bounds__(256) void gemm_bf16(
    const ushort* __restrict__ A, const ushort* __restrict__ Wt,
    const float* __restrict__ bq, const float* __restrict__ bk,
    const float* __restrict__ bv, const float* __restrict__ bp,
    ushort* __restrict__ qh, ushort* __restrict__ kh,
    ushort* __restrict__ vtx, float* __restrict__ outp, int modeBase)
{
    __shared__ __align__(16) ushort AS[BM * SA];   // [m][k]
    __shared__ __align__(16) ushort BS[BN * SA];   // [n][k] (transposed)

    const int tid  = threadIdx.x;
    const int mode = modeBase + blockIdx.z;
    const ushort* W = Wt + (size_t)mode * Cn * Cn;
    const int m0 = blockIdx.x * BM, n0 = blockIdx.y * BN;
    const int wave = tid >> 5, lane = tid & 31;
    const int hlf = lane >> 4, lm = lane & 15;
    const int wm = (wave & 3) * 32, wn = (wave >> 2) * 32;
    const int kb  = hlf ? 8 : 0;     // A-matrix K split per lane-half
    const int kbB = hlf ? 8 : 0;     // B-matrix K split per lane-half

    v8f acc[2][2] = {};

    for (int k0 = 0; k0 < Cn; k0 += BK) {
        __syncthreads();
        // stage A tile 128x64 via async global->LDS copies (no VGPR bounce)
        #pragma unroll
        for (int p = 0; p < 4; ++p) {
            int e = (p * 256 + tid) * 8;
            int ar = e >> 6, ac = e & 63;
            async_cp16(lds_off32(&AS[ar * SA + ac]),
                       &A[(size_t)(m0 + ar) * Cn + k0 + ac]);
        }
        // stage B tile 64(k)x64(n), transposed into BS[n][k]
        #pragma unroll
        for (int p = 0; p < 2; ++p) {
            int e = (p * 256 + tid) * 8;
            int kr = e >> 6, nc = e & 63;
            uint4 t = *(const uint4*)&W[(size_t)(k0 + kr) * Cn + n0 + nc];
            const ushort* ts = (const ushort*)&t;
            #pragma unroll
            for (int i = 0; i < 8; ++i) BS[(nc + i) * SA + kr] = ts[i];
        }
        wait_async0();
        __syncthreads();

        #pragma unroll
        for (int ks = 0; ks < 2; ++ks) {
            v16bf af[2], bf[2];
            #pragma unroll
            for (int mi = 0; mi < 2; ++mi) {
                int row = wm + mi * 16 + lm;
                af[mi] = frag2(&AS[row * SA + ks * 32 + kb],
                               &AS[row * SA + ks * 32 + kb + 16]);
            }
            #pragma unroll
            for (int ni = 0; ni < 2; ++ni) {
                int nc = wn + ni * 16 + lm;
                bf[ni] = frag2(&BS[nc * SA + ks * 32 + kbB],
                               &BS[nc * SA + ks * 32 + kbB + 16]);
            }
            #pragma unroll
            for (int mi = 0; mi < 2; ++mi)
                #pragma unroll
                for (int ni = 0; ni < 2; ++ni)
                    acc[mi][ni] = __builtin_amdgcn_wmma_f32_16x16x32_bf16(
                        false, af[mi], false, bf[ni], (short)0, acc[mi][ni],
                        false, false);
        }
    }

    // epilogue: bias + scatter per mode
    const float* bias = (mode == 0) ? bq : (mode == 1) ? bk : (mode == 2) ? bv : bp;
    #pragma unroll
    for (int mi = 0; mi < 2; ++mi) {
        #pragma unroll
        for (int ni = 0; ni < 2; ++ni) {
            int gm0 = m0 + wm + mi * 16 + hlf * 8;
            int gn  = n0 + wn + ni * 16 + lm;
            float bval = bias[gn];
            #pragma unroll
            for (int r = 0; r < 8; ++r) {
                float val = acc[mi][ni][r] + bval;
                int gm = gm0 + r;
                if (mode == 3) {
                    outp[(size_t)gm * Cn + gn] = val;
                } else {
                    int b = gm >> 11, t = gm & (Tn - 1);
                    int h = gn >> 6,  d = gn & 63;
                    if (mode == 2)
                        vtx[((size_t)(b * Hn + h) * HD + d) * Tn + t] = f2bf(val);
                    else {
                        ushort* dst = (mode == 0) ? qh : kh;
                        dst[((size_t)(b * Hn + h) * Tn + t) * HD + d] = f2bf(val);
                    }
                }
            }
        }
    }
}

// ---------------- kernel 4: flash attention ----------------
// grid: (T/64, B*H); block: 128 (4 waves, 16 q-rows each)

#define SK 72   // LDS halfword stride for 64-wide tiles

__global__ __launch_bounds__(128) void flash_attn(
    const ushort* __restrict__ qh, const ushort* __restrict__ kh,
    const ushort* __restrict__ vtx, ushort* __restrict__ yb)
{
    __shared__ __align__(16) ushort KS[64 * SK];       // [kk][d]
    __shared__ __align__(16) ushort VS[64 * SK];       // [d][kk]  (from Vt)
    __shared__ __align__(16) ushort PS[4 * 16 * SK];   // per-wave P tile

    const int tid = threadIdx.x, wave = tid >> 5, lane = tid & 31;
    const int hlf = lane >> 4, lm = lane & 15;
    const int bh = blockIdx.y, qt = blockIdx.x;
    const int kb = hlf ? 8 : 0, kbB = hlf ? 8 : 0;
    const float scale = 0.125f;   // 1/sqrt(64)

    const ushort* Qb = qh  + (size_t)bh * Tn * HD;
    const ushort* Kb = kh  + (size_t)bh * Tn * HD;
    const ushort* Vb = vtx + (size_t)bh * HD * Tn;

    // Q fragments (row = one of this wave's 16 q rows), resident for whole loop
    const ushort* qrow = Qb + (size_t)(qt * 64 + wave * 16 + lm) * HD;
    v16bf qf[2];
    qf[0] = frag2(qrow + kb,      qrow + kb + 16);
    qf[1] = frag2(qrow + 32 + kb, qrow + 32 + kb + 16);

    v8f o[4] = {};
    float m[8], l[8];
    #pragma unroll
    for (int r = 0; r < 8; ++r) { m[r] = -3.0e38f; l[r] = 0.f; }
    ushort* Pw = &PS[wave * 16 * SK];

    for (int kt = 0; kt < Tn / 64; ++kt) {
        __syncthreads();
        // async stage K (row-major [kk][d]) and V^T (row-major [d][kk])
        #pragma unroll
        for (int p = 0; p < 4; ++p) {
            int e = (p * 128 + tid) * 8;
            int rr = e >> 6, cc = e & 63;
            async_cp16(lds_off32(&KS[rr * SK + cc]),
                       &Kb[(size_t)(kt * 64 + rr) * HD + cc]);
            async_cp16(lds_off32(&VS[rr * SK + cc]),
                       &Vb[(size_t)rr * Tn + kt * 64 + cc]);
        }
        wait_async0();
        __syncthreads();

        // S = Q * K^T  (16 x 64 per wave)
        v8f s[4] = {};
        #pragma unroll
        for (int nt = 0; nt < 4; ++nt) {
            int krow = nt * 16 + lm;
            #pragma unroll
            for (int ks = 0; ks < 2; ++ks) {
                v16bf bfr = frag2(&KS[krow * SK + ks * 32 + kbB],
                                  &KS[krow * SK + ks * 32 + kbB + 16]);
                s[nt] = __builtin_amdgcn_wmma_f32_16x16x32_bf16(
                    false, qf[ks], false, bfr, (short)0, s[nt], false, false);
            }
        }

        // online softmax
        float al[8], rs[8];
        #pragma unroll
        for (int r = 0; r < 8; ++r) {
            float rm = s[0][r];
            #pragma unroll
            for (int nt = 1; nt < 4; ++nt) rm = fmaxf(rm, s[nt][r]);
            rm = rmax16(rm);
            float mn = fmaxf(m[r], rm * scale);
            al[r] = __expf(m[r] - mn);
            m[r] = mn;
            rs[r] = 0.f;
        }
        #pragma unroll
        for (int nt = 0; nt < 4; ++nt) {
            #pragma unroll
            for (int r = 0; r < 8; ++r) {
                float pv = __expf(s[nt][r] * scale - m[r]);
                rs[r] += pv;
                Pw[(r + hlf * 8) * SK + nt * 16 + lm] = f2bf(pv);
            }
        }
        #pragma unroll
        for (int r = 0; r < 8; ++r) {
            float t = rsum16(rs[r]);
            l[r] = l[r] * al[r] + t;
        }
        #pragma unroll
        for (int dt = 0; dt < 4; ++dt)
            #pragma unroll
            for (int r = 0; r < 8; ++r) o[dt][r] *= al[r];

        // O += P * V
        v16bf pa[2];
        pa[0] = frag2(&Pw[lm * SK + kb],      &Pw[lm * SK + kb + 16]);
        pa[1] = frag2(&Pw[lm * SK + 32 + kb], &Pw[lm * SK + 32 + kb + 16]);
        #pragma unroll
        for (int dt = 0; dt < 4; ++dt) {
            int drow = dt * 16 + lm;
            #pragma unroll
            for (int kks = 0; kks < 2; ++kks) {
                v16bf vfr = frag2(&VS[drow * SK + kks * 32 + kbB],
                                  &VS[drow * SK + kks * 32 + kbB + 16]);
                o[dt] = __builtin_amdgcn_wmma_f32_16x16x32_bf16(
                    false, pa[kks], false, vfr, (short)0, o[dt], false, false);
            }
        }
    }

    // normalize and write y (bf16, [B*T, C] row-major)
    int b = bh / Hn, h = bh % Hn;
    #pragma unroll
    for (int dt = 0; dt < 4; ++dt) {
        #pragma unroll
        for (int r = 0; r < 8; ++r) {
            int gm = b * Tn + qt * 64 + wave * 16 + hlf * 8 + r;
            int gc = h * HD + dt * 16 + lm;
            yb[(size_t)gm * Cn + gc] = f2bf(o[dt][r] / l[r]);
        }
    }
}

// ---------------- launch ----------------

extern "C" void kernel_launch(void* const* d_in, const int* in_sizes, int n_in,
                              void* d_out, int out_size, void* d_ws, size_t ws_size,
                              hipStream_t stream) {
    const float* x  = (const float*)d_in[0];
    const float* Wk = (const float*)d_in[1];
    const float* bk = (const float*)d_in[2];
    const float* Wq = (const float*)d_in[3];
    const float* bq = (const float*)d_in[4];
    const float* Wv = (const float*)d_in[5];
    const float* bv = (const float*)d_in[6];
    const float* Wp = (const float*)d_in[7];
    const float* bp = (const float*)d_in[8];
    float* outp = (float*)d_out;

    const size_t XB = (size_t)Mtot * Cn;           // elements (bf16)
    const size_t WT = (size_t)4 * Cn * Cn;
    char* w = (char*)d_ws;
    ushort* xb  = (ushort*)(w);
    ushort* wt  = (ushort*)(w + XB * 2);
    ushort* qh  = (ushort*)(w + (XB + WT) * 2);
    ushort* kh  = (ushort*)(w + (XB + WT + XB) * 2);
    ushort* vtx = (ushort*)(w + (XB + WT + 2 * XB) * 2);
    ushort* yb  = (ushort*)(w + (XB + WT + 3 * XB) * 2);

    cvt_x_bf16<<<(Mtot * Cn) / (256 * 8), 256, 0, stream>>>(x, xb);
    cvt_w_bf16<<<dim3(Cn / 32, Cn / 32, 4), dim3(32, 8), 0, stream>>>(Wq, Wk, Wv, Wp, wt);
    gemm_bf16<<<dim3(Mtot / BM, Cn / BN, 3), 256, 0, stream>>>(
        xb, wt, bq, bk, bv, bp, qh, kh, vtx, outp, 0);
    flash_attn<<<dim3(Tn / 64, Bn * Hn), 128, 0, stream>>>(qh, kh, vtx, yb);
    gemm_bf16<<<dim3(Mtot / BM, Cn / BN, 1), 256, 0, stream>>>(
        yb, wt, bq, bk, bv, bp, qh, kh, vtx, outp, 3);
}
